// CovLayer_23416161698166
// MI455X (gfx1250) — matrix operand order repaired and checked by reading, not compiled, not run on hardware
//
#include <hip/hip_runtime.h>
#include <hip/hip_bf16.h>

typedef __attribute__((ext_vector_type(16))) _Float16 v16h;
typedef __attribute__((ext_vector_type(8)))  _Float16 v8h;
typedef __attribute__((ext_vector_type(8)))  float    v8f;

#define N_MSA 128
#define RRES  256
#define EEMB  32
#define DEMB  128

#define SO_STRIDE 1032        // O tile rows: 1024 k + 8 pad halfs (2064B, 16B aligned)

static __device__ __forceinline__ v16h cat8(v8h a, v8h b) {
  return __builtin_shufflevector(a, b, 0,1,2,3,4,5,6,7,8,9,10,11,12,13,14,15);
}

// ---------------------------------------------------------------------------
// Prep: dst[c*R + r] = (f16) src[r*C + c]   (transpose + fp32->fp16)
//  xl (R=128 n, C=8192 il) -> xlt[i*4096 + l*128 + n]
//  xr (R=128 n, C=8192 jr) -> xrt[j*4096 + r*128 + n]
//  W  (R=1024 k, C=128  d) -> Wt [d*1024 + k]
// dst writes are coalesced; src reads hit L2 after first touch.
// ---------------------------------------------------------------------------
__global__ __launch_bounds__(256)
void transpose_cvt_kernel(const float* __restrict__ src, _Float16* __restrict__ dst,
                          int R, int C) {
  size_t e = (size_t)blockIdx.x * 256 + threadIdx.x;
  if (e >= (size_t)R * C) return;
  int r = (int)(e % R);
  int c = (int)(e / R);
  dst[e] = (_Float16)src[(size_t)r * C + c];
}

// ---------------------------------------------------------------------------
// Main fused kernel: pre-converted f16 inputs in fragment-native layouts.
// Block = 256 threads (8 waves): one i, 16 j's. Phase A: no LDS, fragments
// straight from L2. Phase B: O redistributed through LDS, W fragments from L2.
// ---------------------------------------------------------------------------
__global__ __launch_bounds__(256)
void opm_main_kernel(const _Float16* __restrict__ xlt,  // [i][l][n]
                     const _Float16* __restrict__ xrt,  // [j][r][n]
                     const _Float16* __restrict__ Wt,   // [d][k]
                     const float* __restrict__ bias,
                     float* __restrict__ out)
{
  __shared__ __align__(16) _Float16 sO[16 * SO_STRIDE];

  const int tid  = threadIdx.x;
  const int lane = tid & 31;
  const int wv   = tid >> 5;          // 0..7
  const int hi   = (lane >> 4) & 1;   // lane group
  const int ln   = lane & 15;
  const int i    = blockIdx.x;
  const int j0   = blockIdx.y * 16;

  // ---------------- Phase A: O_p(32x32) = xl_i^T @ xr_p over n=128 ----------
  v8f acc1[2][2][2];                  // [pair][l-tile][r-tile]
  #pragma unroll
  for (int a = 0; a < 2; ++a)
    #pragma unroll
    for (int b = 0; b < 2; ++b)
      #pragma unroll
      for (int cc = 0; cc < 2; ++cc) { v8f z = {}; acc1[a][b][cc] = z; }

  const _Float16* xli = xlt + (size_t)i * (EEMB * N_MSA);
  #pragma unroll
  for (int c = 0; c < 4; ++c) {       // K (=n) chunks of 32
    const int n0 = c * 32;
    // A fragments (16x32 f16): lanes<16 hold K 0-7,16-23; lanes>=16 K 8-15,24-31
    v16h af[2];
    #pragma unroll
    for (int lt = 0; lt < 2; ++lt) {
      const _Float16* ab = xli + (size_t)(lt * 16 + ln) * N_MSA + n0 + hi * 8;
      af[lt] = cat8(*(const v8h*)ab, *(const v8h*)(ab + 16));
    }
    // B fragments (32x16): lanes<16 hold K 0-7,16-23; lanes>=16 K 8-15,24-31, N=ln
    #pragma unroll
    for (int pp = 0; pp < 2; ++pp) {
      int p = wv * 2 + pp;
      const _Float16* xrp = xrt + (size_t)(j0 + p) * (EEMB * N_MSA);
      #pragma unroll
      for (int rt = 0; rt < 2; ++rt) {
        const _Float16* bb = xrp + (size_t)(rt * 16 + ln) * N_MSA + n0 + hi * 8;
        v16h bf = cat8(*(const v8h*)bb, *(const v8h*)(bb + 16));
        acc1[pp][0][rt] = __builtin_amdgcn_wmma_f32_16x16x32_f16(
            false, af[0], false, bf, (short)0, acc1[pp][0][rt], false, false);
        acc1[pp][1][rt] = __builtin_amdgcn_wmma_f32_16x16x32_f16(
            false, af[1], false, bf, (short)0, acc1[pp][1][rt], false, false);
      }
    }
  }

  // ---------------- Spill O to LDS as stage-2 A operand (f16) ---------------
  #pragma unroll
  for (int pp = 0; pp < 2; ++pp) {
    int p = wv * 2 + pp;
    #pragma unroll
    for (int lt = 0; lt < 2; ++lt)
      #pragma unroll
      for (int rt = 0; rt < 2; ++rt)
        #pragma unroll
        for (int g = 0; g < 8; ++g) {
          int l = lt * 16 + g + hi * 8;   // C/D layout: M = g (+8 high lanes)
          int r = rt * 16 + ln;           // N = lane
          sO[p * SO_STRIDE + l * 32 + r] = (_Float16)acc1[pp][lt][rt][g];
        }
  }
  __syncthreads();

  // ---------------- Phase B: Out(16x128) = O(16x1024) @ W(1024x128) ---------
  v8f acc2 = {};
  const int dcol = wv * 16 + ln;          // each wave owns a 16-wide d tile
  const _Float16* wrow = Wt + (size_t)dcol * 1024;
  #pragma unroll 4
  for (int kc = 0; kc < 32; ++kc) {
    const int k0 = kc * 32;
    const _Float16* ab = &sO[ln * SO_STRIDE + k0 + hi * 8];
    v16h a2 = cat8(*(const v8h*)ab, *(const v8h*)(ab + 16));
    const _Float16* bb = wrow + k0 + hi * 8;
    v16h b2 = cat8(*(const v8h*)bb, *(const v8h*)(bb + 16));
    acc2 = __builtin_amdgcn_wmma_f32_16x16x32_f16(
        false, a2, false, b2, (short)0, acc2, false, false);
  }

  // ---------------- Epilogue: bias + store ----------------------------------
  float bv = bias[dcol];
  #pragma unroll
  for (int g = 0; g < 8; ++g) {
    int p = g + hi * 8;
    size_t off = ((size_t)i * RRES + (size_t)(j0 + p)) * DEMB + dcol;
    out[off] = acc2[g] + bv;
  }
}

// ---------------------------------------------------------------------------
// Fallback fused kernel (self-contained, converts in-kernel) for small d_ws.
// ---------------------------------------------------------------------------
#define SA_STRIDE 40
#define SB_JSTRIDE (32 * 40)
#define SW_STRIDE 40

__global__ __launch_bounds__(256)
void opm_fused_fallback(const float* __restrict__ xl, const float* __restrict__ xr,
                        const float* __restrict__ W,  const float* __restrict__ bias,
                        float* __restrict__ out)
{
  __shared__ __align__(16) union {
    struct { _Float16 A[32 * SA_STRIDE]; _Float16 B[16 * SB_JSTRIDE]; } s1;
    struct { _Float16 O[16 * SO_STRIDE]; _Float16 Wt[128 * SW_STRIDE]; } s2;
  } sm;

  const int tid  = threadIdx.x;
  const int lane = tid & 31;
  const int wv   = tid >> 5;
  const int hi   = (lane >> 4) & 1;
  const int ln   = lane & 15;
  const int i    = blockIdx.x;
  const int j0   = blockIdx.y * 16;

  v8f acc1[2][2][2];
  #pragma unroll
  for (int a = 0; a < 2; ++a)
    #pragma unroll
    for (int b = 0; b < 2; ++b)
      #pragma unroll
      for (int cc = 0; cc < 2; ++cc) { v8f z = {}; acc1[a][b][cc] = z; }

  for (int c = 0; c < 4; ++c) {
    const int n0 = c * 32;
    __syncthreads();
    #pragma unroll
    for (int t = 0; t < 4; ++t) {
      int e  = tid + t * 256;
      int nn = e >> 5, l = e & 31;
      float v = xl[(size_t)(n0 + nn) * (RRES * EEMB) + (size_t)i * EEMB + l];
      sm.s1.A[l * SA_STRIDE + nn] = (_Float16)v;
    }
    #pragma unroll
    for (int t = 0; t < 64; ++t) {
      int e   = tid + t * 256;
      int nn  = e >> 9;
      int rem = e & 511;
      int jj  = rem >> 5, r = rem & 31;
      float v = xr[(size_t)(n0 + nn) * (RRES * EEMB) + (size_t)(j0 + jj) * EEMB + r];
      sm.s1.B[jj * SB_JSTRIDE + r * SA_STRIDE + nn] = (_Float16)v;
    }
    __syncthreads();

    v16h af[2];
    #pragma unroll
    for (int lt = 0; lt < 2; ++lt) {
      int l  = lt * 16 + ln;
      int nb = hi * 8;
      af[lt] = cat8(*(const v8h*)&sm.s1.A[l * SA_STRIDE + nb],
                    *(const v8h*)&sm.s1.A[l * SA_STRIDE + nb + 16]);
    }
    #pragma unroll
    for (int pp = 0; pp < 2; ++pp) {
      int p = wv * 2 + pp;
      #pragma unroll
      for (int rt = 0; rt < 2; ++rt) {
        const _Float16* base = &sm.s1.B[p * SB_JSTRIDE + (rt * 16 + ln) * SA_STRIDE + hi * 8];
        v16h bf = cat8(*(const v8h*)base, *(const v8h*)(base + 16));
        acc1[pp][0][rt] = __builtin_amdgcn_wmma_f32_16x16x32_f16(
            false, af[0], false, bf, (short)0, acc1[pp][0][rt], false, false);
        acc1[pp][1][rt] = __builtin_amdgcn_wmma_f32_16x16x32_f16(
            false, af[1], false, bf, (short)0, acc1[pp][1][rt], false, false);
      }
    }
  }

  __syncthreads();
  #pragma unroll
  for (int pp = 0; pp < 2; ++pp) {
    int p = wv * 2 + pp;
    #pragma unroll
    for (int lt = 0; lt < 2; ++lt)
      #pragma unroll
      for (int rt = 0; rt < 2; ++rt)
        #pragma unroll
        for (int g = 0; g < 8; ++g) {
          int l = lt * 16 + g + hi * 8;
          int r = rt * 16 + ln;
          sm.s2.O[p * SO_STRIDE + l * 32 + r] = (_Float16)acc1[pp][lt][rt][g];
        }
  }

  v8f acc2 = {};
  const int dcol = wv * 16 + ln;
  for (int kc = 0; kc < 32; ++kc) {
    const int k0 = kc * 32;
    __syncthreads();
    #pragma unroll
    for (int t = 0; t < 16; ++t) {
      int e  = tid + t * 256;
      int kk = e >> 7, d = e & 127;
      sm.s2.Wt[d * SW_STRIDE + kk] = (_Float16)W[(size_t)(k0 + kk) * DEMB + d];
    }
    __syncthreads();
    const _Float16* ab = &sm.s2.O[ln * SO_STRIDE + k0 + hi * 8];
    v16h a2 = cat8(*(const v8h*)ab, *(const v8h*)(ab + 16));
    const _Float16* bb = &sm.s2.Wt[dcol * SW_STRIDE + hi * 8];
    v16h b2 = cat8(*(const v8h*)bb, *(const v8h*)(bb + 16));
    acc2 = __builtin_amdgcn_wmma_f32_16x16x32_f16(
        false, a2, false, b2, (short)0, acc2, false, false);
  }

  float bv = bias[dcol];
  #pragma unroll
  for (int g = 0; g < 8; ++g) {
    int p = g + hi * 8;
    size_t off = ((size_t)i * RRES + (size_t)(j0 + p)) * DEMB + dcol;
    out[off] = acc2[g] + bv;
  }
}

// ---------------------------------------------------------------------------
extern "C" void kernel_launch(void* const* d_in, const int* in_sizes, int n_in,
                              void* d_out, int out_size, void* d_ws, size_t ws_size,
                              hipStream_t stream) {
  (void)in_sizes; (void)n_in; (void)out_size;
  const float* xl   = (const float*)d_in[0];
  const float* xr   = (const float*)d_in[1];
  const float* W    = (const float*)d_in[2];
  const float* bias = (const float*)d_in[3];
  float* out = (float*)d_out;

  const size_t n_xl = (size_t)N_MSA * RRES * EEMB;   // 1,048,576
  const size_t n_w  = (size_t)EEMB * EEMB * DEMB;    // 131,072
  const size_t need = (2 * n_xl + n_w) * sizeof(_Float16);  // ~4.25 MB

  dim3 grid(RRES, RRES / 16);
  if (ws_size >= need) {
    _Float16* xlt = (_Float16*)d_ws;
    _Float16* xrt = xlt + n_xl;
    _Float16* Wt  = xrt + n_xl;
    // one-time (per launch) transpose+convert into fragment-native layouts
    transpose_cvt_kernel<<<(int)((n_xl + 255) / 256), 256, 0, stream>>>(
        xl, xlt, N_MSA, RRES * EEMB);
    transpose_cvt_kernel<<<(int)((n_xl + 255) / 256), 256, 0, stream>>>(
        xr, xrt, N_MSA, RRES * EEMB);
    transpose_cvt_kernel<<<(int)((n_w + 255) / 256), 256, 0, stream>>>(
        W, Wt, EEMB * EEMB, DEMB);
    opm_main_kernel<<<grid, 256, 0, stream>>>(xlt, xrt, Wt, bias, out);
  } else {
    opm_fused_fallback<<<grid, 256, 0, stream>>>(xl, xr, W, bias, out);
  }
}
